// Encoder_25572235281053
// MI455X (gfx1250) — compile-verified, matrix operand order checked by reading, not run on hardware
//
#include <hip/hip_runtime.h>
#include <hip/hip_bf16.h>
#include <math.h>

// AttentiveFP forward for MI455X (gfx1250, wave32).
// Dense layers -> v_wmma_f32_16x16x32_f16. Weights pre-packed to f16 with
// K padded to 32 (zero-filled) => branch-free, conversion-free B operand and
// no tail-select logic (finite-A x zero-B = 0). Each wave owns a 16x128 C strip;
// all 8 B fragments are loaded (16 b128 in flight) before the 8 WMMAs issue.
// REQUIREMENT (guaranteed by host code): Ncols % 128 == 0.
// Segment softmax -> uint-flip atomicMax + atomicAdd.  Scatters -> float atomics.
//
// Input flattening assumption (setup_inputs() dict insertion order, params
// recursively in insertion order):
//  0 h[N,NF] 1 e[E,EF]
//  gc: 2 node_W 3 node_b 4 e1_W 5 e1_b 6 e2_W 7 e2_b 8 et_W 9 et_b
//      10 gru_Wi 11 gru_Wh 12 gru_bi 13 gru_bh
//  layers l=0,1 (base 14+8l): pe_W pe_b pn_W pn_b gru_Wi gru_Wh gru_bi gru_bh
//  readout t=0,1 (base 30+8t): cl_W cl_b pn_W pn_b gru_Wi gru_Wh gru_bi gru_bh
//  46 src 47 dst 48 node_graph 49 n_graphs

typedef __attribute__((ext_vector_type(16))) _Float16 v16h;
typedef __attribute__((ext_vector_type(8)))  _Float16 v8h;
typedef __attribute__((ext_vector_type(8)))  float    v8f;

#define NTILES 8   // 8 x 16 = 128 output columns per wave

// ------------------------- device helpers -------------------------
__device__ __forceinline__ float apply_act(float v, int act) {
  if (act == 1) return v > 0.f ? v : 0.01f * v;  // LeakyReLU(0.01)
  if (act == 2) return fmaxf(v, 0.f);            // ReLU
  return v;
}
__device__ __forceinline__ unsigned f2ord(float f) {
  unsigned u = __float_as_uint(f);
  return (u & 0x80000000u) ? ~u : (u | 0x80000000u);
}
__device__ __forceinline__ float ord2f(unsigned u) {
  return __uint_as_float((u & 0x80000000u) ? (u & 0x7FFFFFFFu) : ~u);
}
__device__ __forceinline__ float sigm(float x) { return 1.f / (1.f + __expf(-x)); }

// A fragment (f32 source): two contiguous 8-float runs (k = kb..kb+7, kb+16..kb+23).
__device__ __forceinline__ v16h load_fragA(const float* __restrict__ p) {
  v16h f;
#pragma unroll
  for (int j = 0; j < 8; ++j) {
    f[j]     = (_Float16)p[j];
    f[j + 8] = (_Float16)p[j + 16];
  }
  return f;
}
// Clamped A fragment for the padded-K iteration: values finite, B side is zero.
__device__ __forceinline__ v16h load_fragA_clamp(const float* __restrict__ rowp,
                                                 int kbase, int kmax) {
  v16h f;
#pragma unroll
  for (int j = 0; j < 8; ++j) {
    f[j]     = (_Float16)rowp[min(kbase + j, kmax)];
    f[j + 8] = (_Float16)rowp[min(kbase + 16 + j, kmax)];
  }
  return f;
}
// B fragment: pre-packed f16, two aligned 16B vector loads, no conversion.
__device__ __forceinline__ v16h load_fragB(const _Float16* __restrict__ p) {
  v8h lo = *(const v8h*)p;
  v8h hi = *(const v8h*)(p + 16);
  v16h f;
#pragma unroll
  for (int j = 0; j < 8; ++j) { f[j] = lo[j]; f[j + 8] = hi[j]; }
  return f;
}

// Pack W[Nc,K] f32 -> Wp[Nc,Kpad] f16, zero-padded beyond K.
__global__ void pack_w_f16(const float* __restrict__ W, _Float16* __restrict__ out,
                           int Nc, int K, int Kpad) {
  long i = blockIdx.x * (long)blockDim.x + threadIdx.x;
  long total = (long)Nc * Kpad;
  if (i < total) {
    int col = (int)(i / Kpad);
    int k   = (int)(i - (long)col * Kpad);
    out[i] = (k < K) ? (_Float16)W[(long)col * K + k] : (_Float16)0.f;
  }
}

// ------------------------- WMMA GEMM: C = act(A @ W^T + b) -------------------------
// A[M,K] f32 row-major, Wp[Ncols,Kpad] f16 (packed torch Linear weight), C[M,Ncols].
// 4 waves/block on consecutive M tiles; each wave: 16 rows x 128 cols of C.
__global__ void __launch_bounds__(128)
gemm_wmma(const float* __restrict__ A, const _Float16* __restrict__ Wp,
          const float* __restrict__ bias, float* __restrict__ C,
          int M, int Ncols, int K, int Kpad, int act) {
  const int lane  = threadIdx.x & 31;
  const int wave  = threadIdx.x >> 5;
  const int mbase = (blockIdx.x * 4 + wave) * 16;
  const int nbase = blockIdx.y * (NTILES * 16);
  if (mbase >= M) return;                       // wave-uniform
  const int l15 = lane & 15;
  const int kb  = (lane & 16) ? 8 : 0;          // ISA 7.12.2 16-bit A layout
  const float*    arp = A  + (long)min(mbase + l15, M - 1) * K;
  const _Float16* bp0 = Wp + (long)(nbase + l15) * Kpad;  // tile nt at + nt*16*Kpad

  const v8f z8 = {0.f, 0.f, 0.f, 0.f, 0.f, 0.f, 0.f, 0.f};
  v8f acc[NTILES] = {z8, z8, z8, z8, z8, z8, z8, z8};
  const int kfull = K & ~31;
  for (int k0 = 0; k0 < kfull; k0 += 32) {
    v16h af = load_fragA(arp + k0 + kb);
    v16h bfs[NTILES];
#pragma unroll
    for (int nt = 0; nt < NTILES; ++nt)
      bfs[nt] = load_fragB(bp0 + (long)nt * 16 * Kpad + k0 + kb);
#pragma unroll
    for (int nt = 0; nt < NTILES; ++nt)
      acc[nt] = __builtin_amdgcn_wmma_f32_16x16x32_f16(false, af, false, bfs[nt],
                                                       (short)0, acc[nt], false, false);
  }
  if (kfull < Kpad) {  // single padded iteration; B rows are zero beyond K
    v16h af = load_fragA_clamp(arp, kfull + kb, K - 1);
    v16h bfs[NTILES];
#pragma unroll
    for (int nt = 0; nt < NTILES; ++nt)
      bfs[nt] = load_fragB(bp0 + (long)nt * 16 * Kpad + kfull + kb);
#pragma unroll
    for (int nt = 0; nt < NTILES; ++nt)
      acc[nt] = __builtin_amdgcn_wmma_f32_16x16x32_f16(false, af, false, bfs[nt],
                                                       (short)0, acc[nt], false, false);
  }
  const int  mofs  = (lane & 16) ? 8 : 0;
  const bool fullM = (mbase + 16 <= M);
#pragma unroll
  for (int nt = 0; nt < NTILES; ++nt) {
    const int col = nbase + nt * 16 + l15;
    const float bv = bias ? bias[col] : 0.f;
    if (fullM) {
#pragma unroll
      for (int r = 0; r < 8; ++r)
        C[(long)(mbase + mofs + r) * Ncols + col] = apply_act(acc[nt][r] + bv, act);
    } else {
#pragma unroll
      for (int r = 0; r < 8; ++r) {
        int m = mbase + mofs + r;
        if (m < M) C[(long)m * Ncols + col] = apply_act(acc[nt][r] + bv, act);
      }
    }
  }
}

// Edge GEMM: A row r = concat(h[src[r]][0:NF], e[r][0:EF]); K = NF+EF.
// h/e candidates loaded unconditionally (clamped), selected in registers;
// padded-K region handled by zero rows of the packed B.
__global__ void __launch_bounds__(128)
gemm_wmma_edge(const float* __restrict__ h, const float* __restrict__ e,
               const int* __restrict__ src, const _Float16* __restrict__ Wp,
               const float* __restrict__ bias, float* __restrict__ C,
               int E, int NF, int EF, int Ncols, int Kpad, int act) {
  const int K     = NF + EF;
  const int lane  = threadIdx.x & 31;
  const int wave  = threadIdx.x >> 5;
  const int mbase = (blockIdx.x * 4 + wave) * 16;
  const int nbase = blockIdx.y * (NTILES * 16);
  if (mbase >= E) return;
  const int  l15   = lane & 15;
  const int  kb    = (lane & 16) ? 8 : 0;
  const int  lrowc = min(mbase + l15, E - 1);
  const long hrow  = (long)src[lrowc];
  const float* hp  = h + hrow * NF;
  const float* ep  = e + (long)lrowc * EF;
  const _Float16* bp0 = Wp + (long)(nbase + l15) * Kpad;

  const v8f z8 = {0.f, 0.f, 0.f, 0.f, 0.f, 0.f, 0.f, 0.f};
  v8f acc[NTILES] = {z8, z8, z8, z8, z8, z8, z8, z8};
#pragma unroll 1
  for (int k0 = 0; k0 < Kpad; k0 += 32) {
    v16h af;
#pragma unroll
    for (int j = 0; j < 8; ++j) {
      int k1 = k0 + kb + j, k2 = k0 + kb + 16 + j;
      float h1 = hp[min(k1, NF - 1)];
      float h2 = hp[min(k2, NF - 1)];
      float e1v = ep[min(max(k1 - NF, 0), EF - 1)];
      float e2v = ep[min(max(k2 - NF, 0), EF - 1)];
      af[j]     = (_Float16)((k1 < NF) ? h1 : e1v);   // k >= K -> finite junk x 0
      af[j + 8] = (_Float16)((k2 < NF) ? h2 : e2v);
    }
    v16h bfs[NTILES];
#pragma unroll
    for (int nt = 0; nt < NTILES; ++nt)
      bfs[nt] = load_fragB(bp0 + (long)nt * 16 * Kpad + k0 + kb);
#pragma unroll
    for (int nt = 0; nt < NTILES; ++nt)
      acc[nt] = __builtin_amdgcn_wmma_f32_16x16x32_f16(false, af, false, bfs[nt],
                                                       (short)0, acc[nt], false, false);
  }
  const int  mofs  = (lane & 16) ? 8 : 0;
  const bool fullM = (mbase + 16 <= E);
#pragma unroll
  for (int nt = 0; nt < NTILES; ++nt) {
    const int col = nbase + nt * 16 + l15;
    const float bv = bias ? bias[col] : 0.f;
    if (fullM) {
#pragma unroll
      for (int r = 0; r < 8; ++r)
        C[(long)(mbase + mofs + r) * Ncols + col] = apply_act(acc[nt][r] + bv, act);
    } else {
#pragma unroll
      for (int r = 0; r < 8; ++r) {
        int m = mbase + mofs + r;
        if (m < E) C[(long)m * Ncols + col] = apply_act(acc[nt][r] + bv, act);
      }
    }
  }
}

// --------------- pair logit: out = lrelu(X1[i1].w[:G] + X2[i2].w[G:2G] + b) ---------------
__global__ void __launch_bounds__(256)
pair_logit_kernel(const float* __restrict__ X1, const int* __restrict__ idx1,
                  const float* __restrict__ X2, const int* __restrict__ idx2,
                  const float* __restrict__ w, const float* __restrict__ bptr,
                  float* __restrict__ out, int n, int G, int relu1) {
  int wid  = (int)((blockIdx.x * (long)blockDim.x + threadIdx.x) >> 5);
  int lane = threadIdx.x & 31;
  if (wid >= n) return;
  long r1 = idx1 ? (long)idx1[wid] : (long)wid;
  long r2 = idx2 ? (long)idx2[wid] : (long)wid;
  float s = 0.f;
  for (int c = lane; c < G; c += 32) {
    float x1 = X1[r1 * G + c];
    if (relu1) x1 = fmaxf(x1, 0.f);
    s += x1 * w[c] + X2[r2 * G + c] * w[G + c];
  }
#pragma unroll
  for (int o = 16; o; o >>= 1) s += __shfl_xor(s, o, 32);
  if (lane == 0) {
    float v = s + bptr[0];
    out[wid] = v > 0.f ? v : 0.01f * v;
  }
}

// ------------------------- segment softmax pieces -------------------------
__global__ void fill_u32(unsigned* p, unsigned v, long n) {
  long i = blockIdx.x * (long)blockDim.x + threadIdx.x;
  if (i < n) p[i] = v;
}
__global__ void fill_f32(float* p, float v, long n) {
  long i = blockIdx.x * (long)blockDim.x + threadIdx.x;
  if (i < n) p[i] = v;
}
__global__ void seg_max_kernel(const float* __restrict__ lg, const int* __restrict__ seg,
                               unsigned* __restrict__ smax, int n) {
  int i = blockIdx.x * blockDim.x + threadIdx.x;
  if (i < n) atomicMax(&smax[seg[i]], f2ord(lg[i]));
}
__global__ void seg_expsum_kernel(const float* __restrict__ lg, const int* __restrict__ seg,
                                  const unsigned* __restrict__ smax, float* __restrict__ ex,
                                  float* __restrict__ ssum, int n) {
  int i = blockIdx.x * blockDim.x + threadIdx.x;
  if (i < n) {
    float v = __expf(lg[i] - ord2f(smax[seg[i]]));
    ex[i] = v;
    atomicAdd(&ssum[seg[i]], v);
  }
}

// out[seg[r]] += a_r * feat[gidx?gidx[r]:r];  a_r = ex[r]/ssum[seg[r]] (or 1).
__global__ void __launch_bounds__(128)
seg_scatter_kernel(const float* __restrict__ feat, const int* __restrict__ gidx,
                   const int* __restrict__ seg, const float* __restrict__ ex,
                   const float* __restrict__ ssum, float* __restrict__ out,
                   int n, int G) {
  int r = blockIdx.x;
  if (r >= n) return;
  int  s = seg[r];
  float a = ex ? (ex[r] / ssum[s]) : 1.f;
  long fr = gidx ? (long)gidx[r] : (long)r;
  for (int c = threadIdx.x; c < G; c += blockDim.x)
    atomicAdd(&out[(long)s * G + c], a * feat[fr * G + c]);
}

__global__ void elu_kernel(float* p, long n) {
  long i = blockIdx.x * (long)blockDim.x + threadIdx.x;
  if (i < n) { float v = p[i]; p[i] = v > 0.f ? v : __expf(v) - 1.f; }
}
__global__ void relu_kernel(float* p, long n) {
  long i = blockIdx.x * (long)blockDim.x + threadIdx.x;
  if (i < n) p[i] = fmaxf(p[i], 0.f);
}

// torch GRUCell elementwise; gi/gh are [rows,3G] from the WMMA GEMMs.
__global__ void __launch_bounds__(128)
gru_kernel(const float* __restrict__ gi, const float* __restrict__ gh,
           const float* __restrict__ hprev, float* __restrict__ out,
           int rows, int G, int relu_out) {
  int row = blockIdx.x;
  if (row >= rows) return;
  const float* pi = gi + (long)row * 3 * G;
  const float* ph = gh + (long)row * 3 * G;
  for (int c = threadIdx.x; c < G; c += blockDim.x) {
    float r  = sigm(pi[c] + ph[c]);
    float z  = sigm(pi[G + c] + ph[G + c]);
    float nn = tanhf(pi[2 * G + c] + r * ph[2 * G + c]);
    float o  = (1.f - z) * nn + z * hprev[(long)row * G + c];
    if (relu_out) o = fmaxf(o, 0.f);
    out[(long)row * G + c] = o;
  }
}

// ------------------------- host orchestration -------------------------
static inline int pad32(int k) { return (k + 31) & ~31; }

extern "C" void kernel_launch(void* const* d_in, const int* in_sizes, int n_in,
                              void* d_out, int out_size, void* d_ws, size_t ws_size,
                              hipStream_t stream) {
  const float* h    = (const float*)d_in[0];
  const float* e    = (const float*)d_in[1];
  const float* node_W = (const float*)d_in[2];
  const float* node_b = (const float*)d_in[3];
  const float* e1_W   = (const float*)d_in[4];
  const float* e1_b   = (const float*)d_in[5];
  const float* e2_W   = (const float*)d_in[6];
  const float* e2_b   = (const float*)d_in[7];
  const float* et_W   = (const float*)d_in[8];
  const float* et_b   = (const float*)d_in[9];
  const float* gc_Wi  = (const float*)d_in[10];
  const float* gc_Wh  = (const float*)d_in[11];
  const float* gc_bi  = (const float*)d_in[12];
  const float* gc_bh  = (const float*)d_in[13];
  const int* src        = (const int*)d_in[46];
  const int* dst        = (const int*)d_in[47];
  const int* node_graph = (const int*)d_in[48];

  const int N  = in_sizes[48];
  const int E  = in_sizes[46];
  const int G  = in_sizes[3];          // 128 (multiple of 128 required)
  const int NF = in_sizes[0] / N;
  const int EF = in_sizes[1] / E;
  const int NG = out_size / G;
  const int KNFp = pad32(NF);          // node_W packed K
  const int KEp  = pad32(NF + EF);     // e1_W packed K
  const int KGp  = pad32(G);           // 128

  // ---- workspace carve-out ----
  size_t off = 0;
  auto carveb = [&](size_t bytes) {
    void* p = (char*)d_ws + off;
    off += (bytes + 255) & ~(size_t)255;
    return p;
  };
  auto carve  = [&](long nfl) { return (float*)carveb((size_t)nfl * 4); };
  auto carveh = [&](long nh)  { return (_Float16*)carveb((size_t)nh * 2); };

  float*    hv_new = carve((long)N * G);
  float*    nf     = carve((long)N * G);
  float*    ctx    = carve((long)N * G);
  float*    hvtmp  = carve((long)N * G);
  float*    gi     = carve((long)N * 3 * G);
  float*    gh     = carve((long)N * 3 * G);
  float*    he1    = carve((long)E * G);
  float*    et     = carve((long)E * G);
  float*    logits = carve(E);
  float*    ex     = carve(E);
  unsigned* smax   = (unsigned*)carve(N);
  float*    ssum   = carve(N);
  float*    gf     = carve((long)NG * G);
  float*    gtmp   = carve((long)NG * G);
  float*    ggi    = carve((long)NG * 3 * G);
  float*    ggh    = carve((long)NG * 3 * G);
  // packed f16 weights
  _Float16* wnode  = carveh((long)G * KNFp);
  _Float16* we1    = carveh((long)G * KEp);
  _Float16* wet    = carveh((long)G * KGp);
  _Float16* wgcWi  = carveh((long)3 * G * KGp);
  _Float16* wgcWh  = carveh((long)3 * G * KGp);
  _Float16* wpnL[2], *wWiL[2], *wWhL[2], *wpnR[2], *wWiR[2], *wWhR[2];
  for (int l = 0; l < 2; ++l) {
    wpnL[l] = carveh((long)G * KGp);
    wWiL[l] = carveh((long)3 * G * KGp);
    wWhL[l] = carveh((long)3 * G * KGp);
  }
  for (int t = 0; t < 2; ++t) {
    wpnR[t] = carveh((long)G * KGp);
    wWiR[t] = carveh((long)3 * G * KGp);
    wWhR[t] = carveh((long)3 * G * KGp);
  }
  (void)ws_size; (void)n_in;

  auto pack = [&](const float* W, _Float16* dst, int Nc, int K, int Kp) {
    long total = (long)Nc * Kp;
    pack_w_f16<<<(unsigned)((total + 255) / 256), 256, 0, stream>>>(W, dst, Nc, K, Kp);
  };
  // ---- pack all weights (deterministic, every call) ----
  pack(node_W, wnode, G, NF, KNFp);
  pack(e1_W,   we1,   G, NF + EF, KEp);
  pack(et_W,   wet,   G, G, KGp);
  pack(gc_Wi,  wgcWi, 3 * G, G, KGp);
  pack(gc_Wh,  wgcWh, 3 * G, G, KGp);
  for (int l = 0; l < 2; ++l) {
    const int b = 14 + 8 * l;
    pack((const float*)d_in[b + 2], wpnL[l], G, G, KGp);
    pack((const float*)d_in[b + 4], wWiL[l], 3 * G, G, KGp);
    pack((const float*)d_in[b + 5], wWhL[l], 3 * G, G, KGp);
  }
  for (int t = 0; t < 2; ++t) {
    const int b = 30 + 8 * t;
    pack((const float*)d_in[b + 2], wpnR[t], G, G, KGp);
    pack((const float*)d_in[b + 4], wWiR[t], 3 * G, G, KGp);
    pack((const float*)d_in[b + 5], wWhR[t], 3 * G, G, KGp);
  }

  auto gemm = [&](const float* A, const _Float16* Wp, const float* b, float* C,
                  int M, int Nc, int K, int Kp, int act) {
    dim3 grd((M + 63) / 64, Nc / 128);          // Nc is 128 or 384
    gemm_wmma<<<grd, 128, 0, stream>>>(A, Wp, b, C, M, Nc, K, Kp, act);
  };
  auto zero = [&](float* p, long n) {
    fill_f32<<<(unsigned)((n + 255) / 256), 256, 0, stream>>>(p, 0.f, n);
  };
  auto softmax = [&](const float* lg, const int* seg, int n_items, int n_seg) {
    fill_u32<<<(unsigned)((n_seg + 255) / 256), 256, 0, stream>>>(smax, 0u, n_seg);
    fill_f32<<<(unsigned)((n_seg + 255) / 256), 256, 0, stream>>>(ssum, 0.f, n_seg);
    seg_max_kernel<<<(n_items + 255) / 256, 256, 0, stream>>>(lg, seg, smax, n_items);
    seg_expsum_kernel<<<(n_items + 255) / 256, 256, 0, stream>>>(lg, seg, smax, ex, ssum, n_items);
  };

  // ============ GetContext ============
  gemm(h, wnode, node_b, hv_new, N, G, NF, KNFp, /*lrelu*/1);
  {
    dim3 grd((E + 63) / 64, G / 128);
    gemm_wmma_edge<<<grd, 128, 0, stream>>>(h, e, src, we1, e1_b, he1,
                                            E, NF, EF, G, KEp, 1);
  }
  pair_logit_kernel<<<(E + 7) / 8, 256, 0, stream>>>(hv_new, dst, he1, nullptr,
                                                     e2_W, e2_b, logits, E, G, 0);
  softmax(logits, dst, E, N);
  gemm(he1, wet, et_b, et, E, G, G, KGp, 0);
  zero(ctx, (long)N * G);
  seg_scatter_kernel<<<E, 128, 0, stream>>>(et, nullptr, dst, ex, ssum, ctx, E, G);
  elu_kernel<<<(unsigned)(((long)N * G + 255) / 256), 256, 0, stream>>>(ctx, (long)N * G);
  gemm(ctx, wgcWi, gc_bi, gi, N, 3 * G, G, KGp, 0);
  gemm(hv_new, wgcWh, gc_bh, gh, N, 3 * G, G, KGp, 0);
  gru_kernel<<<N, 128, 0, stream>>>(gi, gh, hv_new, nf, N, G, /*relu*/1);

  // ============ GNN layers (R-1 = 2) ============
  for (int l = 0; l < 2; ++l) {
    const int b = 14 + 8 * l;
    const float* pe_W = (const float*)d_in[b + 0];
    const float* pe_b = (const float*)d_in[b + 1];
    const float* pn_b = (const float*)d_in[b + 3];
    const float* l_bi = (const float*)d_in[b + 6];
    const float* l_bh = (const float*)d_in[b + 7];
    pair_logit_kernel<<<(E + 7) / 8, 256, 0, stream>>>(nf, dst, nf, src,
                                                       pe_W, pe_b, logits, E, G, 0);
    softmax(logits, dst, E, N);
    gemm(nf, wpnL[l], pn_b, hvtmp, N, G, G, KGp, 0);
    zero(ctx, (long)N * G);
    seg_scatter_kernel<<<E, 128, 0, stream>>>(hvtmp, src, dst, ex, ssum, ctx, E, G);
    elu_kernel<<<(unsigned)(((long)N * G + 255) / 256), 256, 0, stream>>>(ctx, (long)N * G);
    gemm(ctx, wWiL[l], l_bi, gi, N, 3 * G, G, KGp, 0);
    gemm(nf,  wWhL[l], l_bh, gh, N, 3 * G, G, KGp, 0);
    gru_kernel<<<N, 128, 0, stream>>>(gi, gh, nf, nf, N, G, /*relu*/1);
  }
  // Encoder-level relu(nf) is idempotent after the GRU relu — skipped.

  // ============ Readout (T = 2) ============
  zero(gf, (long)NG * G);
  seg_scatter_kernel<<<N, 128, 0, stream>>>(nf, nullptr, node_graph, nullptr, nullptr, gf, N, G);
  for (int t = 0; t < 2; ++t) {
    const int b = 30 + 8 * t;
    const float* cl_W = (const float*)d_in[b + 0];
    const float* cl_b = (const float*)d_in[b + 1];
    const float* pn_b = (const float*)d_in[b + 3];
    const float* r_bi = (const float*)d_in[b + 6];
    const float* r_bh = (const float*)d_in[b + 7];
    pair_logit_kernel<<<(N + 7) / 8, 256, 0, stream>>>(gf, node_graph, nf, nullptr,
                                                       cl_W, cl_b, logits, N, G, /*relu1*/1);
    softmax(logits, node_graph, N, NG);
    gemm(nf, wpnR[t], pn_b, hvtmp, N, G, G, KGp, 0);
    zero(gtmp, (long)NG * G);
    seg_scatter_kernel<<<N, 128, 0, stream>>>(hvtmp, nullptr, node_graph, ex, ssum, gtmp, N, G);
    // relu(elu(x)) == relu(x): feed relu(segment_sum) straight into the GRU.
    relu_kernel<<<(unsigned)(((long)NG * G + 255) / 256), 256, 0, stream>>>(gtmp, (long)NG * G);
    gemm(gtmp, wWiR[t], r_bi, ggi, NG, 3 * G, G, KGp, 0);
    gemm(gf,   wWhR[t], r_bh, ggh, NG, 3 * G, G, KGp, 0);
    float* gout = (t == 1) ? (float*)d_out : gf;
    gru_kernel<<<NG, 128, 0, stream>>>(ggi, ggh, gf, gout, NG, G, /*relu*/0);
  }
}
